// AsyncLongContextAttention_58463094833569
// MI455X (gfx1250) — matrix operand order repaired and checked by reading, not compiled, not run on hardware
//
#include <hip/hip_runtime.h>
#include <math.h>

typedef __attribute__((ext_vector_type(16))) _Float16 v16h;
typedef __attribute__((ext_vector_type(8)))  _Float16 v8h;
typedef __attribute__((ext_vector_type(4)))  _Float16 v4h;
typedef __attribute__((ext_vector_type(2)))  _Float16 v2h;
typedef __attribute__((ext_vector_type(8)))  float    v8f;

namespace {
constexpr int kB = 2, kS = 2048, kH = 16, kD = 64;
constexpr int kKvTile = 32;                 // KV rows per mainloop step
constexpr int kWaves = 8;                   // waves per workgroup
constexpr int kQRowsPerWave = 16;           // one 16-row Q tile per wave
constexpr int kQRowsPerWG = kWaves * kQRowsPerWave;   // 128
constexpr int kNIter = kS / kKvTile;        // 64
constexpr float kLog2e = 1.44269504088896340736f;
constexpr float kScale = 0.125f;            // 1/sqrt(64)
}

__global__ __launch_bounds__(kWaves * 32, 1)
void fa_fwd_wmma3(const float* __restrict__ Q, const float* __restrict__ K,
                  const float* __restrict__ V, float* __restrict__ O) {
  __shared__ __attribute__((aligned(32))) _Float16 ldsK[2][kKvTile * kD];   // [kv][d]
  __shared__ __attribute__((aligned(32))) _Float16 ldsVT[2][kD * kKvTile];  // [d][kv]
  __shared__ __attribute__((aligned(32))) float    ldsStats[kWaves][16];

  const int tid  = threadIdx.x;
  const int lane = tid & 31;
  const int wv   = tid >> 5;
  const int n16  = lane & 15;   // q column owned by this lane
  const int hi   = lane >> 4;

  const int qblocks = kS / kQRowsPerWG;
  int idx = blockIdx.x;
  const int qb = idx % qblocks; idx /= qblocks;
  const int h  = idx % kH;      idx /= kH;
  const int b  = idx;

  const int q0 = qb * kQRowsPerWG + wv * kQRowsPerWave;
  const int rowStride = kH * kD;

  // ---- Q as B-operand fragments for S^T = K_tile x Q^T (scale*log2e folded) ----
  v16h qbf[2];
  {
    const float* qrow = Q + ((size_t)(b * kS + q0 + n16) * kH + h) * kD;
    const float f = kScale * kLog2e;
    #pragma unroll
    for (int w = 0; w < 2; ++w) {
      v16h bf;
      const int d0 = 32 * w + 16 * hi;
      #pragma unroll
      for (int i = 0; i < 8; ++i) {
        auto p = __builtin_amdgcn_cvt_pkrtz(qrow[d0 + 2*i] * f, qrow[d0 + 2*i + 1] * f);
        bf[2*i] = p[0]; bf[2*i + 1] = p[1];
      }
      qbf[w] = bf;
    }
  }

  v8f acc[4] = {};                          // out rows q = i+8*hi, cols d = 16ch+n16
  float mcur = -INFINITY, lcur = 0.0f;      // per-lane stats for q = n16

  const float* Kbase = K + ((size_t)(b * kS) * kH + h) * kD;
  const float* Vbase = V + ((size_t)(b * kS) * kH + h) * kD;

  // cooperative staging: each thread owns 2 kv rows x 4 d values
  const int sJ  = (tid >> 4) * 2;           // 0,2,..,30
  const int sD0 = (tid & 15) * 4;           // 0..60 (16 threads cover 256B/row)
  const size_t sOffA = (size_t)sJ * rowStride + sD0;
  const size_t sOffB = (size_t)(sJ + 1) * rowStride + sD0;

  float kA[4], kB[4], vA[4], vB[4];
  auto loadTile = [&](int it2) {
    const size_t off = (size_t)it2 * kKvTile * rowStride;
    const float* ka = Kbase + off + sOffA;
    const float* kb = Kbase + off + sOffB;
    const float* va = Vbase + off + sOffA;
    const float* vb = Vbase + off + sOffB;
    #pragma unroll
    for (int i = 0; i < 4; ++i) { kA[i] = ka[i]; kB[i] = kb[i]; vA[i] = va[i]; vB[i] = vb[i]; }
  };
  auto storeTile = [&](int buf) {
    v4h kh0, kh1;
    {
      auto a = __builtin_amdgcn_cvt_pkrtz(kA[0], kA[1]);
      auto c = __builtin_amdgcn_cvt_pkrtz(kA[2], kA[3]);
      kh0[0] = a[0]; kh0[1] = a[1]; kh0[2] = c[0]; kh0[3] = c[1];
      auto d = __builtin_amdgcn_cvt_pkrtz(kB[0], kB[1]);
      auto e = __builtin_amdgcn_cvt_pkrtz(kB[2], kB[3]);
      kh1[0] = d[0]; kh1[1] = d[1]; kh1[2] = e[0]; kh1[3] = e[1];
    }
    *(v4h*)&ldsK[buf][sJ * kD + sD0]       = kh0;   // ds_store_b64
    *(v4h*)&ldsK[buf][(sJ + 1) * kD + sD0] = kh1;   // ds_store_b64
    #pragma unroll
    for (int i = 0; i < 4; ++i) {                   // half2 {V[j][d], V[j+1][d]}
      auto p = __builtin_amdgcn_cvt_pkrtz(vA[i], vB[i]);
      v2h pv; pv[0] = p[0]; pv[1] = p[1];
      *(v2h*)&ldsVT[buf][(sD0 + i) * kKvTile + sJ] = pv;   // ds_store_b32
    }
  };

  loadTile(0);
  storeTile(0);
  __syncthreads();

  for (int it = 0; it < kNIter; ++it) {
    const int cur = it & 1;
    const bool havenext = (it + 1 < kNIter);
    if (havenext) loadTile(it + 1);       // waitcnt sinks past the compute

    // ---- preload all K A-fragments, then 4 score WMMAs (pipelined dscnt) ----
    v8h afl[2][2], afh[2][2];
    #pragma unroll
    for (int t = 0; t < 2; ++t)
      #pragma unroll
      for (int w = 0; w < 2; ++w) {
        afl[t][w] = *(const v8h*)&ldsK[cur][(n16 + 16 * t) * kD + 32 * w + 16 * hi];
        afh[t][w] = *(const v8h*)&ldsK[cur][(n16 + 16 * t) * kD + 32 * w + 16 * hi + 8];
      }
    v8f s[2];
    #pragma unroll
    for (int t = 0; t < 2; ++t) {
      v8f c = {};
      #pragma unroll
      for (int w = 0; w < 2; ++w) {
        v16h af;
        #pragma unroll
        for (int i = 0; i < 8; ++i) { af[i] = afl[t][w][i]; af[8 + i] = afh[t][w][i]; }
        c = __builtin_amdgcn_wmma_f32_16x16x32_f16(false, af, false, qbf[w],
                                                   (short)0, c, false, false);
      }
      s[t] = c;
    }

    // issue V B-fragment loads now; latency hides under the softmax VALU below
    v16h vbf[4];
    #pragma unroll
    for (int ch = 0; ch < 4; ++ch)
      vbf[ch] = __builtin_shufflevector(*(const v8h*)&ldsVT[cur][(16 * ch + n16) * kKvTile + 8 * hi],
                                        *(const v8h*)&ldsVT[cur][(16 * ch + n16) * kKvTile + 16 + 8 * hi],
                                        0, 1, 2, 3, 4, 5, 6, 7, 8, 9, 10, 11, 12, 13, 14, 15);

    // ---- online softmax: in-lane reduce over kv, one cross-half combine ----
    float mx = fmaxf(s[0][0], s[1][0]);
    #pragma unroll
    for (int i = 1; i < 8; ++i) mx = fmaxf(mx, fmaxf(s[0][i], s[1][i]));
    mx = fmaxf(mx, __shfl_xor(mx, 16, 32));
    const float mprev = mcur;
    mcur = fmaxf(mprev, mx);
    const float cf = __builtin_amdgcn_exp2f(mprev - mcur);   // raw v_exp_f32
    float p0[8], p1[8], rs = 0.0f;
    #pragma unroll
    for (int i = 0; i < 8; ++i) {
      p0[i] = __builtin_amdgcn_exp2f(s[0][i] - mcur);
      p1[i] = __builtin_amdgcn_exp2f(s[1][i] - mcur);
      rs += p0[i] + p1[i];
    }
    rs += __shfl_xor(rs, 16, 32);
    lcur = lcur * cf + rs;

    // rescale only when some lane's max actually grew (wave-uniform branch)
    if (__any(mcur > mprev)) {
      ldsStats[wv][n16] = cf;
      const v8f cfrow = *(const v8f*)&ldsStats[wv][8 * hi];
      #pragma unroll
      for (int ch = 0; ch < 4; ++ch)
        #pragma unroll
        for (int i = 0; i < 8; ++i)
          acc[ch][i] *= cfrow[i];
    }

    // ---- P A-fragment directly from transposed probabilities (packed cvt) ----
    v16h pa;
    #pragma unroll
    for (int i = 0; i < 4; ++i) {
      auto a = __builtin_amdgcn_cvt_pkrtz(p0[2*i], p0[2*i + 1]);
      auto c = __builtin_amdgcn_cvt_pkrtz(p1[2*i], p1[2*i + 1]);
      pa[2*i] = a[0]; pa[2*i + 1] = a[1];
      pa[8 + 2*i] = c[0]; pa[8 + 2*i + 1] = c[1];
    }

    #pragma unroll
    for (int ch = 0; ch < 4; ++ch)
      acc[ch] = __builtin_amdgcn_wmma_f32_16x16x32_f16(false, pa, false, vbf[ch],
                                                       (short)0, acc[ch], false, false);

    if (havenext) storeTile(cur ^ 1);     // other buffer: nobody reads it yet
    __syncthreads();                      // single barrier per iteration
  }

  // ---- epilogue: broadcast l per q, normalize, coalesced fp32 store ----
  ldsStats[wv][n16] = lcur;
  const v8f lrow = *(const v8f*)&ldsStats[wv][8 * hi];
  #pragma unroll
  for (int i = 0; i < 8; ++i) {
    const float inv_l = 1.0f / lrow[i];
    float* orow = O + ((size_t)(b * kS + q0 + i + 8 * hi) * kH + h) * kD;
    #pragma unroll
    for (int ch = 0; ch < 4; ++ch)
      orow[16 * ch + n16] = acc[ch][i] * inv_l;
  }
}

extern "C" void kernel_launch(void* const* d_in, const int* in_sizes, int n_in,
                              void* d_out, int out_size, void* d_ws, size_t ws_size,
                              hipStream_t stream) {
  const float* Q = (const float*)d_in[0];
  const float* K = (const float*)d_in[1];
  const float* V = (const float*)d_in[2];
  float* O = (float*)d_out;
  (void)in_sizes; (void)n_in; (void)out_size; (void)d_ws; (void)ws_size;
  const int nblocks = kB * kH * (kS / kQRowsPerWG);   // 512
  fa_fwd_wmma3<<<nblocks, kWaves * 32, 0, stream>>>(Q, K, V, O);
}
